// MOE_86543591014909
// MI455X (gfx1250) — compile-verified, matrix operand order checked by reading, not run on hardware
//
#include <hip/hip_runtime.h>
#include <hip/hip_bf16.h>
#include <math.h>

// Problem constants (from reference)
#define C_DIM 768
#define E_NUM 8
#define H_DIM 3072
#define N_TOK 2048
#define NPAIR (N_TOK * 2)      // total expert-row assignments (top-2)

// GEMM tiling
#define BM 64                  // rows per workgroup tile (4 waves x 16)
#define BN 64                  // cols per workgroup tile
#define BK 32                  // K per WMMA step
#define LDS_STRIDE 40          // padded bf16 row stride for transposed B tile (80 B = 10 u64)
#define LDS_ROW64 10           // u64 words per LDS row
#define LDS_BUF (BN * LDS_STRIDE)        // elements per buffer
#define LDS_BUF64 (LDS_BUF / 4)          // u64 words per buffer
#define MT_MAX (N_TOK / BM)    // worst-case m-tiles per expert = 32

typedef __attribute__((ext_vector_type(16))) __bf16 bf16x16;
typedef __attribute__((ext_vector_type(2)))  __bf16 bf16x2;
typedef __attribute__((ext_vector_type(8)))  float  f32x8;

// ---- workspace layout (bytes) ----
#define WS_CNT   0
#define WS_BASE  256
#define WS_EARR  512
#define WS_SARR  (WS_EARR + NPAIR * 4)
#define WS_PARR  (WS_SARR + NPAIR * 4)
#define WS_XE    65536
#define WS_HBUF  (WS_XE   + (size_t)NPAIR * C_DIM * 2)   // bf16 Xe
#define WS_OBUF  (WS_HBUF + (size_t)NPAIR * H_DIM * 2)   // bf16 Hbuf
// Obuf: NPAIR * C_DIM f32  -> total ~44 MB

__device__ inline bf16x16 load_bf16x16(const __bf16* p) {
    union { bf16x16 v; uint4 u[2]; } r;
    const uint4* q = (const uint4*)p;
    r.u[0] = q[0];
    r.u[1] = q[1];
    return r.v;
}

// pack two f32 -> packed bf16 pair; ext-vector inserts match v_cvt_pk_bf16_f32
__device__ inline unsigned pack_bf16(float a, float b) {
    bf16x2 v;
    v.x = (__bf16)a;             // low half  = lower K
    v.y = (__bf16)b;             // high half = upper K
    union { bf16x2 h; unsigned u; } r;
    r.h = v;
    return r.u;
}

#define WMMA_BF16(a, b, c) __builtin_amdgcn_wmma_f32_16x16x32_bf16( \
    false, (a), false, (b), (short)0, (c), false, false)

// ---------------- kernel 1: zero expert counters ----------------
__global__ void moe_init(int* cnt) {
    if (threadIdx.x < E_NUM) cnt[threadIdx.x] = 0;
}

// ---------------- kernel 2: router (logits, top-2, probs, slots) ----------------
__global__ __launch_bounds__(256)
void moe_router(const float* __restrict__ x, const float* __restrict__ wr,
                int* cnt, int* eArr, int* sArr, float* pArr) {
    __shared__ float swr[C_DIM * E_NUM];     // 24 KB
    __shared__ float slog[32 * E_NUM];
    for (int i = threadIdx.x; i < C_DIM * E_NUM; i += blockDim.x) swr[i] = wr[i];
    __syncthreads();

    int tLocal = threadIdx.x >> 3;           // 0..31
    int e      = threadIdx.x & 7;            // expert handled by this lane
    int t      = blockIdx.x * 32 + tLocal;

    const float* xr = x + (size_t)t * C_DIM;
    float acc = 0.0f;
    #pragma unroll 4
    for (int c = 0; c < C_DIM; ++c) acc += xr[c] * swr[c * E_NUM + e];
    slog[tLocal * E_NUM + e] = acc;
    __syncthreads();

    if (e == 0) {
        const float* lg = &slog[tLocal * E_NUM];
        int be = 0; float bv = lg[0];
        #pragma unroll
        for (int k = 1; k < E_NUM; ++k) if (lg[k] > bv) { bv = lg[k]; be = k; }
        int se = (be == 0) ? 1 : 0; float sv = lg[se];
        #pragma unroll
        for (int k = 0; k < E_NUM; ++k)
            if (k != be && lg[k] > sv) { sv = lg[k]; se = k; }
        float e1 = __expf(sv - bv);
        float inv = 1.0f / (1.0f + e1);
        float p0 = inv, p1 = e1 * inv;
        int s0 = atomicAdd(&cnt[be], 1);
        int s1 = atomicAdd(&cnt[se], 1);
        eArr[2 * t]     = be;  eArr[2 * t + 1] = se;
        sArr[2 * t]     = s0;  sArr[2 * t + 1] = s1;
        pArr[2 * t]     = p0;  pArr[2 * t + 1] = p1;
    }
}

// ---------------- kernel 3: exclusive prefix sum over 8 counts ----------------
__global__ void moe_prefix(const int* cnt, int* base) {
    if (threadIdx.x == 0) {
        int a = 0;
        for (int e = 0; e < E_NUM; ++e) { base[e] = a; a += cnt[e]; }
    }
}

// ---------------- kernel 4: gather token rows into compacted bf16 buffer ----------------
__global__ __launch_bounds__(256)
void moe_gather(const float* __restrict__ x, const int* __restrict__ eArr,
                const int* __restrict__ sArr, const int* __restrict__ base,
                __bf16* __restrict__ Xe) {
    int pair = blockIdx.x;                   // 0..NPAIR-1
    int t = pair >> 1;
    int row = base[eArr[pair]] + sArr[pair];
    const float* src = x + (size_t)t * C_DIM;
    __bf16* dst = Xe + (size_t)row * C_DIM;
    for (int i = threadIdx.x; i < C_DIM; i += 256)
        dst[i] = (__bf16)src[i];
}

// ---------------- kernel 5: fused fc + gate GEMM + SwiGLU ----------------
// grid: (E_NUM*MT_MAX, H_DIM/BN), block 128 (4 waves), wave tile 16x64.
// Double-buffered LDS; A fragment + B tiles software-pipelined across K;
// B fragments rotated (distance 2) so ds_load latency hides under WMMA.
__global__ __launch_bounds__(128)
void moe_mlp1(const __bf16* __restrict__ Xe,
              const float* __restrict__ Wf, const float* __restrict__ Wg,
              const float* __restrict__ bF, const float* __restrict__ bG,
              const int* __restrict__ cnt, const int* __restrict__ base,
              __bf16* __restrict__ Hbuf) {
    __shared__ __align__(16) __bf16 sBf[2 * LDS_BUF];
    __shared__ __align__(16) __bf16 sBg[2 * LDS_BUF];

    int e  = blockIdx.x / MT_MAX;
    int mt = blockIdx.x % MT_MAX;
    int cntE = cnt[e];
    int m0 = mt * BM;
    if (m0 >= cntE) return;
    int baseE = base[e];
    int nBase = blockIdx.y * BN;
    const float* wf = Wf + (size_t)e * C_DIM * H_DIM;
    const float* wg = Wg + (size_t)e * C_DIM * H_DIM;

    int tid  = threadIdx.x;
    int wave = tid >> 5;
    int lane = tid & 31;
    int nloc = lane & 15;
    int kh   = (lane >> 4) << 4;             // lane half owns K-half (0 or 16)

    int mCl = m0 + wave * 16 + nloc;
    if (mCl >= cntE) mCl = cntE - 1;
    const __bf16* aRow = Xe + (size_t)(baseE + mCl) * C_DIM + kh;

    // staging geometry: thread owns K-quad (4 consecutive K rows) of one float4 column
    int kq = tid >> 4;                       // 0..7 -> K rows 4*kq..4*kq+3
    int c4 = tid & 15;                       // float4 column
    const float* gF = wf + (size_t)(4 * kq) * H_DIM + nBase + c4 * 4;
    const float* gG = wg + (size_t)(4 * kq) * H_DIM + nBase + c4 * 4;
    uint2* sF64 = (uint2*)sBf;
    uint2* sG64 = (uint2*)sBg;
    int sW = (c4 * 4) * LDS_ROW64 + kq;      // u64 index of (n = 4*c4, k-quad kq)

    // prologue: K-tile 0 staged into registers; A fragment for k0=0
    float4 rF[4], rG[4];
    #pragma unroll
    for (int r = 0; r < 4; ++r) {
        rF[r] = *(const float4*)(gF + (size_t)r * H_DIM);
        rG[r] = *(const float4*)(gG + (size_t)r * H_DIM);
    }
    bf16x16 aCur = load_bf16x16(aRow);

    f32x8 zf = {};
    f32x8 accF[4], accG[4];
    #pragma unroll
    for (int i = 0; i < 4; ++i) { accF[i] = zf; accG[i] = zf; }

    int buf = 0;
    for (int k0 = 0; k0 < C_DIM; k0 += BK) {
        int bo   = buf * LDS_BUF;
        int bo64 = buf * LDS_BUF64;
        // commit staged registers to LDS: packed bf16 pairs, b64 stores
        #pragma unroll
        for (int j = 0; j < 4; ++j) {
            const float* f = (const float*)rF;
            const float* g = (const float*)rG;
            uint2 vF, vG;
            vF.x = pack_bf16(f[0 * 4 + j], f[1 * 4 + j]);
            vF.y = pack_bf16(f[2 * 4 + j], f[3 * 4 + j]);
            vG.x = pack_bf16(g[0 * 4 + j], g[1 * 4 + j]);
            vG.y = pack_bf16(g[2 * 4 + j], g[3 * 4 + j]);
            sF64[bo64 + sW + j * LDS_ROW64] = vF;
            sG64[bo64 + sW + j * LDS_ROW64] = vG;
        }
        __syncthreads();
        // pipeline next K-tile: global loads land while this tile's WMMAs run
        bf16x16 aNxt = aCur;
        if (k0 + BK < C_DIM) {
            size_t off = (size_t)(k0 + BK) * H_DIM;
            #pragma unroll
            for (int r = 0; r < 4; ++r) {
                rF[r] = *(const float4*)(gF + off + (size_t)r * H_DIM);
                rG[r] = *(const float4*)(gG + off + (size_t)r * H_DIM);
            }
            aNxt = load_bf16x16(aRow + k0 + BK);
        }
        // compute: rotated B-fragment pipeline, distance 2
        const __bf16* fb = &sBf[bo + nloc * LDS_STRIDE + kh];
        const __bf16* gb = &sBg[bo + nloc * LDS_STRIDE + kh];
        bf16x16 b0 = load_bf16x16(fb);
        bf16x16 b1 = load_bf16x16(fb + 16 * LDS_STRIDE);
        bf16x16 b2;
        b2 = load_bf16x16(fb + 32 * LDS_STRIDE); accF[0] = WMMA_BF16(aCur, b0, accF[0]);
        b0 = load_bf16x16(fb + 48 * LDS_STRIDE); accF[1] = WMMA_BF16(aCur, b1, accF[1]);
        b1 = load_bf16x16(gb);                   accF[2] = WMMA_BF16(aCur, b2, accF[2]);
        b2 = load_bf16x16(gb + 16 * LDS_STRIDE); accF[3] = WMMA_BF16(aCur, b0, accF[3]);
        b0 = load_bf16x16(gb + 32 * LDS_STRIDE); accG[0] = WMMA_BF16(aCur, b1, accG[0]);
        b1 = load_bf16x16(gb + 48 * LDS_STRIDE); accG[1] = WMMA_BF16(aCur, b2, accG[1]);
        accG[2] = WMMA_BF16(aCur, b0, accG[2]);
        accG[3] = WMMA_BF16(aCur, b1, accG[3]);
        aCur = aNxt;
        buf ^= 1;
    }

    // epilogue: out = (h + bF) * silu(g + bG), stored bf16
    int mHi = (lane >> 4) * 8;
    #pragma unroll
    for (int nt = 0; nt < 4; ++nt) {
        int col = nBase + nt * 16 + nloc;
        float bf_v = bF[(size_t)e * H_DIM + col];
        float bg_v = bG[(size_t)e * H_DIM + col];
        #pragma unroll
        for (int r = 0; r < 8; ++r) {
            int m = wave * 16 + mHi + r;
            if (m0 + m < cntE) {
                float h = accF[nt][r] + bf_v;
                float g = accG[nt][r] + bg_v;
                float s = g / (1.0f + __expf(-g));   // silu(g)
                Hbuf[(size_t)(baseE + m0 + m) * H_DIM + col] = (__bf16)(h * s);
            }
        }
    }
}

// ---------------- kernel 6: down projection GEMM ----------------
__global__ __launch_bounds__(128)
void moe_mlp2(const __bf16* __restrict__ Hbuf,
              const float* __restrict__ Wp, const float* __restrict__ bP,
              const int* __restrict__ cnt, const int* __restrict__ base,
              float* __restrict__ Obuf) {
    __shared__ __align__(16) __bf16 sB[2 * LDS_BUF];

    int e  = blockIdx.x / MT_MAX;
    int mt = blockIdx.x % MT_MAX;
    int cntE = cnt[e];
    int m0 = mt * BM;
    if (m0 >= cntE) return;
    int baseE = base[e];
    int nBase = blockIdx.y * BN;
    const float* wp = Wp + (size_t)e * H_DIM * C_DIM;

    int tid  = threadIdx.x;
    int wave = tid >> 5;
    int lane = tid & 31;
    int nloc = lane & 15;
    int kh   = (lane >> 4) << 4;

    int mCl = m0 + wave * 16 + nloc;
    if (mCl >= cntE) mCl = cntE - 1;
    const __bf16* aRow = Hbuf + (size_t)(baseE + mCl) * H_DIM + kh;

    int kq = tid >> 4;
    int c4 = tid & 15;
    const float* gP = wp + (size_t)(4 * kq) * C_DIM + nBase + c4 * 4;
    uint2* s64 = (uint2*)sB;
    int sW = (c4 * 4) * LDS_ROW64 + kq;

    float4 rP[4];
    #pragma unroll
    for (int r = 0; r < 4; ++r) rP[r] = *(const float4*)(gP + (size_t)r * C_DIM);
    bf16x16 aCur = load_bf16x16(aRow);

    f32x8 zf = {};
    f32x8 acc[4];
    #pragma unroll
    for (int i = 0; i < 4; ++i) acc[i] = zf;

    int buf = 0;
    for (int k0 = 0; k0 < H_DIM; k0 += BK) {
        int bo   = buf * LDS_BUF;
        int bo64 = buf * LDS_BUF64;
        #pragma unroll
        for (int j = 0; j < 4; ++j) {
            const float* f = (const float*)rP;
            uint2 v;
            v.x = pack_bf16(f[0 * 4 + j], f[1 * 4 + j]);
            v.y = pack_bf16(f[2 * 4 + j], f[3 * 4 + j]);
            s64[bo64 + sW + j * LDS_ROW64] = v;
        }
        __syncthreads();
        bf16x16 aNxt = aCur;
        if (k0 + BK < H_DIM) {
            size_t off = (size_t)(k0 + BK) * C_DIM;
            #pragma unroll
            for (int r = 0; r < 4; ++r) rP[r] = *(const float4*)(gP + off + (size_t)r * C_DIM);
            aNxt = load_bf16x16(aRow + k0 + BK);
        }
        const __bf16* pb = &sB[bo + nloc * LDS_STRIDE + kh];
        bf16x16 b0 = load_bf16x16(pb);
        bf16x16 b1 = load_bf16x16(pb + 16 * LDS_STRIDE);
        bf16x16 b2;
        b2 = load_bf16x16(pb + 32 * LDS_STRIDE); acc[0] = WMMA_BF16(aCur, b0, acc[0]);
        b0 = load_bf16x16(pb + 48 * LDS_STRIDE); acc[1] = WMMA_BF16(aCur, b1, acc[1]);
        acc[2] = WMMA_BF16(aCur, b2, acc[2]);
        acc[3] = WMMA_BF16(aCur, b0, acc[3]);
        aCur = aNxt;
        buf ^= 1;
    }

    int mHi = (lane >> 4) * 8;
    #pragma unroll
    for (int nt = 0; nt < 4; ++nt) {
        int col = nBase + nt * 16 + nloc;
        float b = bP[(size_t)e * C_DIM + col];
        #pragma unroll
        for (int r = 0; r < 8; ++r) {
            int m = wave * 16 + mHi + r;
            if (m0 + m < cntE)
                Obuf[(size_t)(baseE + m0 + m) * C_DIM + col] = acc[nt][r] + b;
        }
    }
}

// ---------------- kernel 7: weighted combine back to token order ----------------
__global__ __launch_bounds__(256)
void moe_combine(const float* __restrict__ Obuf,
                 const int* __restrict__ eArr, const int* __restrict__ sArr,
                 const float* __restrict__ pArr, const int* __restrict__ base,
                 float* __restrict__ y) {
    int t = blockIdx.x;
    int r0 = base[eArr[2 * t]]     + sArr[2 * t];
    int r1 = base[eArr[2 * t + 1]] + sArr[2 * t + 1];
    float p0 = pArr[2 * t], p1 = pArr[2 * t + 1];
    const float* o0 = Obuf + (size_t)r0 * C_DIM;
    const float* o1 = Obuf + (size_t)r1 * C_DIM;
    float* yr = y + (size_t)t * C_DIM;
    for (int i = threadIdx.x; i < C_DIM; i += 256)
        yr[i] = p0 * o0[i] + p1 * o1[i];
}

extern "C" void kernel_launch(void* const* d_in, const int* in_sizes, int n_in,
                              void* d_out, int out_size, void* d_ws, size_t ws_size,
                              hipStream_t stream) {
    (void)in_sizes; (void)n_in; (void)out_size; (void)ws_size;
    const float* x   = (const float*)d_in[0];
    const float* wr  = (const float*)d_in[1];
    const float* wf  = (const float*)d_in[2];
    const float* bF  = (const float*)d_in[3];
    const float* wg  = (const float*)d_in[4];
    const float* bG  = (const float*)d_in[5];
    const float* wp  = (const float*)d_in[6];
    const float* bP  = (const float*)d_in[7];
    // d_in[8] = top_k (compile-time 2)

    char* ws = (char*)d_ws;
    int*    cnt  = (int*)(ws + WS_CNT);
    int*    base = (int*)(ws + WS_BASE);
    int*    eArr = (int*)(ws + WS_EARR);
    int*    sArr = (int*)(ws + WS_SARR);
    float*  pArr = (float*)(ws + WS_PARR);
    __bf16* Xe   = (__bf16*)(ws + WS_XE);
    __bf16* Hbuf = (__bf16*)(ws + WS_HBUF);
    float*  Obuf = (float*)(ws + WS_OBUF);
    float*  y    = (float*)d_out;

    moe_init   <<<1, 32, 0, stream>>>(cnt);
    moe_router <<<N_TOK / 32, 256, 0, stream>>>(x, wr, cnt, eArr, sArr, pArr);
    moe_prefix <<<1, 1, 0, stream>>>(cnt, base);
    moe_gather <<<NPAIR, 256, 0, stream>>>(x, eArr, sArr, base, Xe);

    dim3 g1(E_NUM * MT_MAX, H_DIM / BN);
    moe_mlp1   <<<g1, 128, 0, stream>>>(Xe, wf, wg, bF, bG, cnt, base, Hbuf);

    dim3 g2(E_NUM * MT_MAX, C_DIM / BN);
    moe_mlp2   <<<g2, 128, 0, stream>>>(Hbuf, wp, bP, cnt, base, Obuf);

    moe_combine<<<N_TOK, 256, 0, stream>>>(Obuf, eArr, sArr, pArr, base, y);
}
